// EncoderBlock_20693152432730
// MI455X (gfx1250) — matrix operand order repaired and checked by reading, not compiled, not run on hardware
//
#include <hip/hip_runtime.h>
#include <hip/hip_bf16.h>

// ---------------------------------------------------------------------------
// Transformer encoder block for MI455X (gfx1250, wave32, WMMA).
// Compute-bound (~275 GFLOP vs ~10us of HBM traffic at 23.3 TB/s), so all
// GEMMs run on v_wmma_f32_16x16x32_bf16 with fp32 accumulation.
// GEMM tiles staged with GLOBAL_LOAD_ASYNC_TO_LDS_B128 (ASYNCcnt-tracked).
// ---------------------------------------------------------------------------

typedef unsigned short u16;
typedef __bf16 bf16x16 __attribute__((ext_vector_type(16)));
typedef float  f32x8   __attribute__((ext_vector_type(8)));
typedef float  f32x4   __attribute__((ext_vector_type(4)));
typedef unsigned int u32x4 __attribute__((ext_vector_type(4)));
typedef int    v4i    __attribute__((__vector_size__(4 * sizeof(int))));

#define AS1 __attribute__((address_space(1)))
#define AS3 __attribute__((address_space(3)))

#if defined(__has_builtin)
#  if __has_builtin(__builtin_amdgcn_global_load_async_to_lds_b128) && \
      __has_builtin(__builtin_amdgcn_s_wait_asynccnt)
#    define HAVE_ASYNC_LDS 1
#  endif
#endif
#ifndef HAVE_ASYNC_LDS
#  define HAVE_ASYNC_LDS 0
#endif

union Frag16 {            // 32 bytes: one WMMA 16-bit A/B fragment per lane
    bf16x16 bf;
    u16     u[16];
    u32x4   q[2];
};

__device__ __forceinline__ u16 f32_to_bf16(float f) {
    unsigned int u = __builtin_bit_cast(unsigned int, f);
    u = (u + 0x7FFFu + ((u >> 16) & 1u)) >> 16;   // round-to-nearest-even
    return (u16)u;
}

__device__ __forceinline__ f32x8 wmma_bf16(const Frag16& a, const Frag16& b, f32x8 c) {
    return __builtin_amdgcn_wmma_f32_16x16x32_bf16(
        false, a.bf, false, b.bf, (short)0, c, false, false);
}

// problem constants
constexpr int CB   = 4;
constexpr int CS   = 2048;
constexpr int CD   = 1024;
constexpr int CH   = 16;
constexpr int CDK  = 64;
constexpr int CFF  = 4096;
constexpr int NTOK = CB * CS;          // 8192

// ---------------------------------------------------------------------------
// Weight convert + transpose: W[K,N] fp32 -> Wt[N,K] bf16
// ---------------------------------------------------------------------------
__global__ void convert_transpose(const float* __restrict__ W, u16* __restrict__ Wt,
                                  int K, int N) {
    int idx = blockIdx.x * 256 + threadIdx.x;
    int k = idx / N, n = idx % N;
    Wt[(size_t)n * K + k] = f32_to_bf16(W[idx]);
}

// ---------------------------------------------------------------------------
// LayerNorm: one wave32 per token row of 1024, output bf16
// ---------------------------------------------------------------------------
__global__ void ln_kernel(const float* __restrict__ x, const float* __restrict__ g,
                          const float* __restrict__ b, u16* __restrict__ out) {
    int lane = threadIdx.x & 31;
    int row  = (blockIdx.x * blockDim.x + threadIdx.x) >> 5;
    const float* xr = x + (size_t)row * CD;
    float v[32];
    float s = 0.f;
#pragma unroll
    for (int i = 0; i < 32; ++i) { v[i] = xr[i * 32 + lane]; s += v[i]; }
#pragma unroll
    for (int m = 1; m < 32; m <<= 1) s += __shfl_xor(s, m, 32);
    float mu = s * (1.f / CD);
    float var = 0.f;
#pragma unroll
    for (int i = 0; i < 32; ++i) { float d = v[i] - mu; var += d * d; }
#pragma unroll
    for (int m = 1; m < 32; m <<= 1) var += __shfl_xor(var, m, 32);
    float rstd = rsqrtf(var * (1.f / CD) + 1e-5f);
    u16* orow = out + (size_t)row * CD;
#pragma unroll
    for (int i = 0; i < 32; ++i) {
        int c = i * 32 + lane;
        orow[c] = f32_to_bf16((v[i] - mu) * rstd * g[c] + b[c]);
    }
}

// ---------------------------------------------------------------------------
// Tiled bf16 WMMA GEMM:  C[M,N] = A[M,K] @ Bt[N,K]^T, fused epilogues
// Block tile 128x128, 8 waves (4x2), each wave 32x64 (2x4 WMMA subtiles)
// ---------------------------------------------------------------------------
enum { MODE_QK = 0, MODE_VT = 1, MODE_RESID = 2, MODE_GELU = 3 };

template <int MODE>
__global__ __launch_bounds__(256)
void gemm_bf16(const u16* __restrict__ A, const u16* __restrict__ Bt,
               const float* __restrict__ bias, const float* __restrict__ resid,
               void* __restrict__ out, int M, int N, int K) {
    constexpr int MT = 128, NT = 128, KT = 32;
    __shared__ __align__(16) u16 As[MT * KT];
    __shared__ __align__(16) u16 Bs[NT * KT];

    int tid  = threadIdx.x;
    int lane = tid & 31, wave = tid >> 5;
    int wm = wave & 3, wn = wave >> 2;                  // 4 x 2 wave grid
    int m0 = blockIdx.x * MT, n0 = blockIdx.y * NT;
    int g = lane >> 4, l15 = lane & 15;

    f32x8 acc[2][4] = {};

    for (int k0 = 0; k0 < K; k0 += KT) {
        __syncthreads();
        // stage A and B tiles: 128 rows x 32 k each (512 x 16B chunks each,
        // 2 A-chunks + 2 B-chunks per thread)
#pragma unroll
        for (int i = 0; i < 2; ++i) {
            int c = tid + i * 256;
            int r = c >> 2, off = (c & 3) * 8;
#if HAVE_ASYNC_LDS
            __builtin_amdgcn_global_load_async_to_lds_b128(
                (AS1 v4i*)&A[(size_t)(m0 + r) * K + k0 + off],
                (AS3 v4i*)&As[r * KT + off], 0, 0);
            __builtin_amdgcn_global_load_async_to_lds_b128(
                (AS1 v4i*)&Bt[(size_t)(n0 + r) * K + k0 + off],
                (AS3 v4i*)&Bs[r * KT + off], 0, 0);
#else
            *(u32x4*)&As[r * KT + off] =
                *(const u32x4*)&A[(size_t)(m0 + r) * K + k0 + off];
            *(u32x4*)&Bs[r * KT + off] =
                *(const u32x4*)&Bt[(size_t)(n0 + r) * K + k0 + off];
#endif
        }
#if HAVE_ASYNC_LDS
        __builtin_amdgcn_s_wait_asynccnt(0);
#else
        if (k0 + KT < K) {  // prefetch next K tile (global_prefetch_b8)
            __builtin_prefetch(&A[(size_t)(m0 + (tid >> 1)) * K + k0 + KT], 0, 0);
            __builtin_prefetch(&Bt[(size_t)(n0 + (tid >> 1)) * K + k0 + KT], 0, 0);
        }
#endif
        __syncthreads();

        Frag16 a[2], b[4];
#pragma unroll
        for (int mi = 0; mi < 2; ++mi) {
            int r = wm * 32 + mi * 16 + l15;
            a[mi].q[0] = *(const u32x4*)&As[r * KT + g * 8];        // K 0-7 / 8-15
            a[mi].q[1] = *(const u32x4*)&As[r * KT + 16 + g * 8];   // K 16-23 / 24-31
        }
#pragma unroll
        for (int ni = 0; ni < 4; ++ni) {
            int r = wn * 64 + ni * 16 + l15;
            b[ni].q[0] = *(const u32x4*)&Bs[r * KT + g * 8];        // K 0-7 / 8-15
            b[ni].q[1] = *(const u32x4*)&Bs[r * KT + 16 + g * 8];   // K 16-23 / 24-31
        }
#pragma unroll
        for (int mi = 0; mi < 2; ++mi)
#pragma unroll
            for (int ni = 0; ni < 4; ++ni)
                acc[mi][ni] = wmma_bf16(a[mi], b[ni], acc[mi][ni]);
    }

    // epilogue: C layout -> lane holds rows (j + g*8), col l15 per subtile
#pragma unroll
    for (int mi = 0; mi < 2; ++mi)
#pragma unroll
        for (int ni = 0; ni < 4; ++ni) {
            int colg = n0 + wn * 64 + ni * 16 + l15;
            float bv = bias[colg];
#pragma unroll
            for (int j = 0; j < 8; ++j) {
                int rowg = m0 + wm * 32 + mi * 16 + j + g * 8;
                float v = acc[mi][ni][j] + bv;
                if constexpr (MODE == MODE_GELU) {
                    v = 0.5f * v * (1.f + erff(v * 0.70710678118654752f));
                    ((u16*)out)[(size_t)rowg * N + colg] = f32_to_bf16(v);
                } else if constexpr (MODE == MODE_RESID) {
                    ((float*)out)[(size_t)rowg * N + colg] =
                        v + resid[(size_t)rowg * N + colg];
                } else {
                    int bb = rowg >> 11, s = rowg & 2047;   // token -> (batch, seq)
                    int h = colg >> 6, d = colg & 63;       // col   -> (head, dk)
                    if constexpr (MODE == MODE_QK) {        // [B,H,S,dk]
                        ((u16*)out)[(((size_t)(bb * CH + h) * CS) + s) * CDK + d] =
                            f32_to_bf16(v);
                    } else {                                // MODE_VT: [B,H,dk,S]
                        ((u16*)out)[(((size_t)(bb * CH + h) * CDK) + d) * CS + s] =
                            f32_to_bf16(v);
                    }
                }
            }
        }
}

// ---------------------------------------------------------------------------
// Flash attention: one wave per 16 query rows; scores & PV on WMMA,
// online softmax in fp32; P transposed to A-fragment layout through LDS.
// Q,K: [B*H, S, dk] bf16   Vt: [B*H, dk, S] bf16   O: [B*S, D] bf16
// ---------------------------------------------------------------------------
__global__ __launch_bounds__(128)
void attn_kernel(const u16* __restrict__ Q, const u16* __restrict__ Km,
                 const u16* __restrict__ Vt, u16* __restrict__ O) {
    __shared__ __align__(16) float Pls[4][16 * 32];   // per-wave P tile (fp32)
    int lane = threadIdx.x & 31, wave = threadIdx.x >> 5;
    int qt = blockIdx.x & 31;             // S/64 query tiles
    int bh = blockIdx.x >> 5;             // batch*head
    int q0 = qt * 64 + wave * 16;
    int g = lane >> 4, l15 = lane & 15;

    const u16* Qh = Q  + (size_t)bh * CS * CDK;
    const u16* Kh = Km + (size_t)bh * CS * CDK;
    const u16* Vh = Vt + (size_t)bh * CDK * CS;

    // Q fragments (held in registers for the whole pass): dk=64 -> 2 K-steps
    Frag16 qa[2];
#pragma unroll
    for (int kk = 0; kk < 2; ++kk) {
        const u16* rp = Qh + (size_t)(q0 + l15) * CDK + kk * 32;
        qa[kk].q[0] = *(const u32x4*)(rp + g * 8);
        qa[kk].q[1] = *(const u32x4*)(rp + 16 + g * 8);
    }

    f32x8 o[4] = {};
    float m[8], l[8];
#pragma unroll
    for (int j = 0; j < 8; ++j) { m[j] = -1e30f; l[j] = 0.f; }
    const float scale = 0.125f;           // 1/sqrt(64)
    float* P = &Pls[wave][0];

    for (int sk = 0; sk < CS; sk += 32) {
        // scores: Q(16x64) @ K_tile^T(64x32) -> 2 subtiles of 16x16
        f32x8 sacc[2] = {};
#pragma unroll
        for (int ns = 0; ns < 2; ++ns)
#pragma unroll
            for (int kk = 0; kk < 2; ++kk) {
                Frag16 kb;   // column n of K^T = row (sk+ns*16+n) of K (contiguous)
                const u16* rp = Kh + (size_t)(sk + ns * 16 + l15) * CDK + kk * 32;
                kb.q[0] = *(const u32x4*)(rp + g * 8);
                kb.q[1] = *(const u32x4*)(rp + 16 + g * 8);
                sacc[ns] = wmma_bf16(qa[kk], kb, sacc[ns]);
            }

        // online softmax over the 32 new columns
#pragma unroll
        for (int j = 0; j < 8; ++j) {
            float s0 = sacc[0][j] * scale, s1 = sacc[1][j] * scale;
            float v = fmaxf(s0, s1);
            v = fmaxf(v, __shfl_xor(v, 1, 32));
            v = fmaxf(v, __shfl_xor(v, 2, 32));
            v = fmaxf(v, __shfl_xor(v, 4, 32));
            v = fmaxf(v, __shfl_xor(v, 8, 32));
            float mn = fmaxf(m[j], v);
            float p0 = __expf(s0 - mn), p1 = __expf(s1 - mn);
            float ps = p0 + p1;
            ps += __shfl_xor(ps, 1, 32);
            ps += __shfl_xor(ps, 2, 32);
            ps += __shfl_xor(ps, 4, 32);
            ps += __shfl_xor(ps, 8, 32);
            float alpha = __expf(m[j] - mn);
            l[j] = l[j] * alpha + ps;
            m[j] = mn;
#pragma unroll
            for (int nd = 0; nd < 4; ++nd) o[nd][j] *= alpha;
            int r = j + g * 8;
            P[r * 32 + l15]      = p0;    // per-wave LDS transpose staging
            P[r * 32 + 16 + l15] = p1;
        }

        // reload P as an A-fragment (row l15, K split per lane half)
        Frag16 pa;
        {
            const float* rp = P + l15 * 32;
            f32x4 f0 = *(const f32x4*)(rp + g * 8);
            f32x4 f1 = *(const f32x4*)(rp + g * 8 + 4);
            f32x4 f2 = *(const f32x4*)(rp + 16 + g * 8);
            f32x4 f3 = *(const f32x4*)(rp + 16 + g * 8 + 4);
#pragma unroll
            for (int i = 0; i < 4; ++i) {
                pa.u[i]      = f32_to_bf16(f0[i]);
                pa.u[4 + i]  = f32_to_bf16(f1[i]);
                pa.u[8 + i]  = f32_to_bf16(f2[i]);
                pa.u[12 + i] = f32_to_bf16(f3[i]);
            }
        }

        // O += P(16x32) @ V_tile(32x64): 4 dk subtiles
#pragma unroll
        for (int nd = 0; nd < 4; ++nd) {
            Frag16 vb;   // column d of V = row d of Vt (contiguous in s)
            const u16* rp = Vh + (size_t)(nd * 16 + l15) * CS + sk + g * 8;
            vb.q[0] = *(const u32x4*)rp;
            vb.q[1] = *(const u32x4*)(rp + 16);
            o[nd] = wmma_bf16(pa, vb, o[nd]);
        }
    }

    // normalize and scatter to token-major [B*S, D] bf16
    int bb = bh >> 4, h = bh & 15;
#pragma unroll
    for (int nd = 0; nd < 4; ++nd)
#pragma unroll
        for (int j = 0; j < 8; ++j) {
            int s = q0 + j + g * 8;
            int d = nd * 16 + l15;
            O[((size_t)(bb * CS + s)) * CD + h * CDK + d] =
                f32_to_bf16(o[nd][j] / l[j]);
        }
}

// ---------------------------------------------------------------------------
// Launch
// ---------------------------------------------------------------------------
extern "C" void kernel_launch(void* const* d_in, const int* in_sizes, int n_in,
                              void* d_out, int out_size, void* d_ws, size_t ws_size,
                              hipStream_t stream) {
    const float* x     = (const float*)d_in[0];
    const float* Wq    = (const float*)d_in[1];
    const float* bq    = (const float*)d_in[2];
    const float* Wk    = (const float*)d_in[3];
    const float* bk    = (const float*)d_in[4];
    const float* Wv    = (const float*)d_in[5];
    const float* bv    = (const float*)d_in[6];
    const float* Wo    = (const float*)d_in[7];
    const float* bo    = (const float*)d_in[8];
    const float* ln1g  = (const float*)d_in[9];
    const float* ln1b  = (const float*)d_in[10];
    const float* ln2g  = (const float*)d_in[11];
    const float* ln2b  = (const float*)d_in[12];
    const float* W1    = (const float*)d_in[13];
    const float* b1    = (const float*)d_in[14];
    const float* W2    = (const float*)d_in[15];
    const float* b2    = (const float*)d_in[16];

    char* ws = (char*)d_ws;
    auto MB = [](size_t m) { return m * 1024 * 1024; };
    u16*  WtQ = (u16*)(ws + MB(0));     // 2 MB each
    u16*  WtK = (u16*)(ws + MB(2));
    u16*  WtV = (u16*)(ws + MB(4));
    u16*  WtO = (u16*)(ws + MB(6));
    u16*  Wt1 = (u16*)(ws + MB(8));     // 8 MB
    u16*  Wt2 = (u16*)(ws + MB(16));    // 8 MB
    u16*  Hbf = (u16*)(ws + MB(24));    // 16 MB  LN1 out
    u16*  Qb  = (u16*)(ws + MB(40));    // 16 MB
    u16*  Kb  = (u16*)(ws + MB(56));    // 16 MB
    u16*  Vtb = (u16*)(ws + MB(72));    // 16 MB
    u16*  Ob  = (u16*)(ws + MB(88));    // 16 MB  attention out
    float* X1 = (float*)(ws + MB(104)); // 32 MB  residual 1 (fp32)
    u16*  H2  = (u16*)(ws + MB(136));   // 16 MB  LN2 out
    u16*  Ab  = (u16*)(ws + MB(152));   // 64 MB  MLP hidden

    // weights -> bf16, transposed
    convert_transpose<<<4096,  256, 0, stream>>>(Wq, WtQ, CD, CD);
    convert_transpose<<<4096,  256, 0, stream>>>(Wk, WtK, CD, CD);
    convert_transpose<<<4096,  256, 0, stream>>>(Wv, WtV, CD, CD);
    convert_transpose<<<4096,  256, 0, stream>>>(Wo, WtO, CD, CD);
    convert_transpose<<<16384, 256, 0, stream>>>(W1, Wt1, CD, CFF);
    convert_transpose<<<16384, 256, 0, stream>>>(W2, Wt2, CFF, CD);

    dim3 gN1024(NTOK / 128, CD / 128);   // (64, 8)
    dim3 gN4096(NTOK / 128, CFF / 128);  // (64, 32)

    // sub-layer 1: LN -> QKV -> attention -> Wo + residual
    ln_kernel<<<NTOK / 8, 256, 0, stream>>>(x, ln1g, ln1b, Hbf);
    gemm_bf16<MODE_QK><<<gN1024, 256, 0, stream>>>(Hbf, WtQ, bq, nullptr, Qb, NTOK, CD, CD);
    gemm_bf16<MODE_QK><<<gN1024, 256, 0, stream>>>(Hbf, WtK, bk, nullptr, Kb, NTOK, CD, CD);
    gemm_bf16<MODE_VT><<<gN1024, 256, 0, stream>>>(Hbf, WtV, bv, nullptr, Vtb, NTOK, CD, CD);
    attn_kernel<<<CB * CH * (CS / 64), 128, 0, stream>>>(Qb, Kb, Vtb, Ob);
    gemm_bf16<MODE_RESID><<<gN1024, 256, 0, stream>>>(Ob, WtO, bo, x, X1, NTOK, CD, CD);

    // sub-layer 2: LN -> W1+GELU -> W2 + residual
    ln_kernel<<<NTOK / 8, 256, 0, stream>>>(X1, ln2g, ln2b, H2);
    gemm_bf16<MODE_GELU><<<gN4096, 256, 0, stream>>>(H2, Wt1, b1, nullptr, Ab, NTOK, CFF, CD);
    gemm_bf16<MODE_RESID><<<gN1024, 256, 0, stream>>>(Ab, Wt2, b2, X1, (float*)d_out, NTOK, CD, CFF);
}
